// AutoCorrelation_88785563943386
// MI455X (gfx1250) — matrix operand order repaired and checked by reading, not compiled, not run on hardware
//
#include <hip/hip_runtime.h>

typedef __attribute__((ext_vector_type(16))) _Float16     v16h;
typedef __attribute__((ext_vector_type(4)))  _Float16     v4h;
typedef __attribute__((ext_vector_type(8)))  float        v8f;
typedef __attribute__((ext_vector_type(4)))  unsigned int v4u;

#define BB    8
#define LL    4096
#define HE    512
#define TOPK  8
#define LMASK 4095

#define QROWS    64    // Q rows per workgroup
#define KROWS    64    // K rows per macro step (f32 fallback kernel)
#define KR2      32    // K rows per step (async double-buffered kernel)
#define LDSTRIDE 520   // 512 + 8 halves pad (breaks LDS bank conflicts)

union Frag16 { v16h v; v4u q[2]; };

__global__ __launch_bounds__(256) void zero_ws(float* __restrict__ C) {
  int gid = blockIdx.x * 256 + threadIdx.x;
  if (gid < BB * LL) C[gid] = 0.f;
}

__global__ __launch_bounds__(256) void cvt_to_f16(const float* __restrict__ s,
                                                  _Float16* __restrict__ d, int n4) {
  int gid = blockIdx.x * 256 + threadIdx.x;
  if (gid < n4) {
    float4 f = ((const float4*)s)[gid];
    v4h o = { (_Float16)f.x, (_Float16)f.y, (_Float16)f.z, (_Float16)f.w };
    ((v4h*)d)[gid] = o;
  }
}

// ---------------------------------------------------------------------------
// Async-DMA path: C[b,l] += sum_t <q[b,t,:], k[b,(t-l)%L,:]> from pre-packed
// f16 Q/K. K blocks are double-buffered via GLOBAL_LOAD_ASYNC_TO_LDS_B128 so
// the DMA for block jb+1 overlaps the WMMA phase for block jb.
// ---------------------------------------------------------------------------
__global__ __launch_bounds__(256) void corr_wmma_async(const _Float16* __restrict__ Qh,
                                                       const _Float16* __restrict__ Kh,
                                                       float* __restrict__ C) {
  __shared__ _Float16 sQ[QROWS * LDSTRIDE];       //  66,560 B
  __shared__ _Float16 sK[2][KR2 * LDSTRIDE];      //  66,560 B (2 buffers)
  __shared__ float    bins[LL];                   //  16,384 B  -> 149,504 B

  const int tid     = threadIdx.x;
  const int batch   = blockIdx.x >> 6;
  const int rowBase = (blockIdx.x & 63) * QROWS;

  const unsigned long long qbase =
      (unsigned long long)(size_t)(Qh + (size_t)batch * LL * HE);
  const unsigned long long kbase =
      (unsigned long long)(size_t)(Kh + (size_t)batch * LL * HE);
  // generic->LDS: low 32 bits of the generic pointer are the LDS byte address
  const unsigned sQ0 = (unsigned)(size_t)&sQ[0];
  const unsigned sK0 = (unsigned)(size_t)&sK[0][0];

  // async-stage Q block: 64 rows x 1024 B (16B per lane-op)
  for (int i = 0; i < 16; ++i) {
    int ci = i * 256 + tid;                      // 0..4095
    int r = ci >> 6, c8 = ci & 63;
    unsigned voff = (unsigned)((rowBase + r) * 1024 + c8 * 16);
    unsigned lds  = sQ0 + (unsigned)((r * LDSTRIDE + c8 * 8) * 2);
    asm volatile("global_load_async_to_lds_b128 %0, %1, %2"
                 :: "v"(lds), "v"(voff), "s"(qbase) : "memory");
  }
  // async-stage K buffer 0: 32 rows x 1024 B
  for (int i = 0; i < 8; ++i) {
    int ci = i * 256 + tid;                      // 0..2047
    int r = ci >> 6, c8 = ci & 63;
    unsigned voff = (unsigned)(r * 1024 + c8 * 16);
    unsigned lds  = sK0 + (unsigned)((r * LDSTRIDE + c8 * 8) * 2);
    asm volatile("global_load_async_to_lds_b128 %0, %1, %2"
                 :: "v"(lds), "v"(voff), "s"(kbase) : "memory");
  }
  for (int l = tid; l < LL; l += 256) bins[l] = 0.f;
  asm volatile("s_wait_asynccnt 0x0" ::: "memory");
  __syncthreads();

  const int wave = tid >> 5, lane = tid & 31;
  const int ri = wave >> 1, cj = wave & 1;       // 4x2 tiles of 16x16
  // A fragment lane params (16-bit A 16x32; this code's contiguous k = 16*(lane/16) + s)
  const int am  = lane & 15;
  const int akb = (lane < 16) ? 0 : 16;
  // B fragment lane params (row-striped 32x16)
  const int bn  = lane & 15;
  const int bkb = (lane < 16) ? 0 : 16;
  const _Float16* qrow = &sQ[(ri * 16 + am) * LDSTRIDE];

  for (int jb = 0; jb < LL / KR2; ++jb) {
    // issue DMA for the NEXT K block into the other buffer (overlaps compute)
    if (jb + 1 < LL / KR2) {
      unsigned dstb = sK0 + (unsigned)(((jb + 1) & 1) * KR2 * LDSTRIDE * 2);
      for (int i = 0; i < 8; ++i) {
        int ci = i * 256 + tid;
        int r = ci >> 6, c8 = ci & 63;
        unsigned voff = (unsigned)(((jb + 1) * KR2 + r) * 1024 + c8 * 16);
        unsigned lds  = dstb + (unsigned)((r * LDSTRIDE + c8 * 8) * 2);
        asm volatile("global_load_async_to_lds_b128 %0, %1, %2"
                     :: "v"(lds), "v"(voff), "s"(kbase) : "memory");
      }
    }

    const _Float16* krow = &sK[jb & 1][(cj * 16 + bn) * LDSTRIDE];
    v8f acc = {};
#pragma unroll
    for (int kc = 0; kc < 16; ++kc) {            // K = 512 in chunks of 32
      const int kof = kc * 32;
      Frag16 a, b0;
      a.q[0]  = *(const v4u*)(qrow + kof + akb);
      a.q[1]  = *(const v4u*)(qrow + kof + akb + 8);
      b0.q[0] = *(const v4u*)(krow + kof + bkb);
      b0.q[1] = *(const v4u*)(krow + kof + bkb + 8);
      acc = __builtin_amdgcn_wmma_f32_16x16x32_f16(false, a.v, false, b0.v,
                                                   (short)0, acc, false, false);
    }

    // fold Gram tile into lag bins: element (t,t') -> lag (t - t') mod L
    const int rowT = rowBase + ri * 16 + ((lane >> 4) * 8);
    const int colT = jb * KR2 + cj * 16 + bn;
#pragma unroll
    for (int vv = 0; vv < 8; ++vv)
      atomicAdd(&bins[(rowT + vv - colT) & LMASK], acc[vv]);

    asm volatile("s_wait_asynccnt 0x0" ::: "memory");
    __syncthreads();
  }

  for (int l = tid; l < LL; l += 256)
    atomicAdd(&C[(size_t)batch * LL + l], bins[l]);
}

// ---------------------------------------------------------------------------
// Fallback (ws too small for the f16 pack): f32 loads + VALU convert staging.
// ---------------------------------------------------------------------------
__global__ __launch_bounds__(256) void corr_wmma_f32(const float* __restrict__ Q,
                                                     const float* __restrict__ K,
                                                     float* __restrict__ C) {
  __shared__ _Float16 sQ[QROWS * LDSTRIDE];
  __shared__ _Float16 sK[KROWS * LDSTRIDE];
  __shared__ float    bins[LL];

  const int tid     = threadIdx.x;
  const int batch   = blockIdx.x >> 6;
  const int rowBase = (blockIdx.x & 63) * QROWS;

  const float* qsrc = Q + ((size_t)batch * LL + rowBase) * HE;
  for (int i = 0; i < 32; ++i) {
    int fi = i * 256 + tid;
    int r = fi >> 7, c4 = fi & 127;
    float4 vv = ((const float4*)qsrc)[(size_t)r * 128 + c4];
    _Float16* d = &sQ[r * LDSTRIDE + c4 * 4];
    d[0] = (_Float16)vv.x; d[1] = (_Float16)vv.y;
    d[2] = (_Float16)vv.z; d[3] = (_Float16)vv.w;
  }
  for (int l = tid; l < LL; l += 256) bins[l] = 0.f;
  __syncthreads();

  const int wave = tid >> 5, lane = tid & 31;
  const int ri  = wave >> 1;
  const int cj0 = (wave & 1) * 2;
  const int am  = lane & 15;
  const int akb = (lane < 16) ? 0 : 16;
  const int bn  = lane & 15;
  const int bkb = (lane < 16) ? 0 : 16;

  const _Float16* qrow  = &sQ[(ri * 16 + am) * LDSTRIDE];
  const _Float16* krow0 = &sK[(cj0 * 16 + bn) * LDSTRIDE];
  const _Float16* krow1 = &sK[((cj0 + 1) * 16 + bn) * LDSTRIDE];

  for (int jb = 0; jb < LL / KROWS; ++jb) {
    const int jBase = jb * KROWS;
    const float* ksrc = K + ((size_t)batch * LL + jBase) * HE;
    for (int i = 0; i < 32; ++i) {
      int fi = i * 256 + tid;
      int r = fi >> 7, c4 = fi & 127;
      float4 vv = ((const float4*)ksrc)[(size_t)r * 128 + c4];
      _Float16* d = &sK[r * LDSTRIDE + c4 * 4];
      d[0] = (_Float16)vv.x; d[1] = (_Float16)vv.y;
      d[2] = (_Float16)vv.z; d[3] = (_Float16)vv.w;
    }
    __syncthreads();

    v8f acc0 = {};
    v8f acc1 = {};
#pragma unroll
    for (int kc = 0; kc < 16; ++kc) {
      const int kof = kc * 32;
      Frag16 a, b0, b1;
      a.q[0]  = *(const v4u*)(qrow  + kof + akb);
      a.q[1]  = *(const v4u*)(qrow  + kof + akb + 8);
      b0.q[0] = *(const v4u*)(krow0 + kof + bkb);
      b0.q[1] = *(const v4u*)(krow0 + kof + bkb + 8);
      b1.q[0] = *(const v4u*)(krow1 + kof + bkb);
      b1.q[1] = *(const v4u*)(krow1 + kof + bkb + 8);
      acc0 = __builtin_amdgcn_wmma_f32_16x16x32_f16(false, a.v, false, b0.v,
                                                    (short)0, acc0, false, false);
      acc1 = __builtin_amdgcn_wmma_f32_16x16x32_f16(false, a.v, false, b1.v,
                                                    (short)0, acc1, false, false);
    }

    const int rowT  = rowBase + ri * 16 + ((lane >> 4) * 8);
    const int colT0 = jBase + cj0 * 16 + bn;
#pragma unroll
    for (int vv = 0; vv < 8; ++vv) {
      atomicAdd(&bins[(rowT + vv - colT0) & LMASK],        acc0[vv]);
      atomicAdd(&bins[(rowT + vv - (colT0 + 16)) & LMASK], acc1[vv]);
    }
    __syncthreads();
  }

  for (int l = tid; l < LL; l += 256)
    atomicAdd(&C[(size_t)batch * LL + l], bins[l]);
}

__global__ __launch_bounds__(256) void topk_softmax(const float* __restrict__ C,
                                                    int* __restrict__ idxOut,
                                                    float* __restrict__ wOut) {
  __shared__ float mv[LL];
  __shared__ float rv[256];
  __shared__ int   ra[256];
  __shared__ int   sidx[TOPK];
  const int tid = threadIdx.x;

  for (int l = tid; l < LL; l += 256) {
    float s = 0.f;
    for (int b = 0; b < BB; ++b) s += C[b * LL + l];
    mv[l] = s;
  }
  __syncthreads();

  for (int it = 0; it < TOPK; ++it) {
    float best = -3.4e38f; int bi = 0;
    for (int l = tid; l < LL; l += 256)
      if (mv[l] > best) { best = mv[l]; bi = l; }
    rv[tid] = best; ra[tid] = bi;
    __syncthreads();
    for (int s = 128; s > 0; s >>= 1) {
      if (tid < s && rv[tid + s] > rv[tid]) { rv[tid] = rv[tid + s]; ra[tid] = ra[tid + s]; }
      __syncthreads();
    }
    if (tid == 0) { sidx[it] = ra[0]; mv[ra[0]] = -3.4e38f; }
    __syncthreads();
  }

  if (tid < TOPK) idxOut[tid] = sidx[tid];
  if (tid < BB) {
    const int b = tid;
    float w[TOPK], mx = -3.4e38f, s = 0.f;
    for (int j = 0; j < TOPK; ++j) {
      w[j] = C[b * LL + sidx[j]] * (1.f / 512.f);
      mx = fmaxf(mx, w[j]);
    }
    for (int j = 0; j < TOPK; ++j) { w[j] = __expf(w[j] - mx); s += w[j]; }
    for (int j = 0; j < TOPK; ++j) wOut[b * TOPK + j] = w[j] / s;
  }
}

__global__ __launch_bounds__(256) void delay_agg(const float* __restrict__ V,
                                                 const float* __restrict__ W,
                                                 const int* __restrict__ IDX,
                                                 float* __restrict__ out) {
  __shared__ int   sIdx[TOPK];
  __shared__ float sW[BB * TOPK];
  const int tid = threadIdx.x;
  if (tid < TOPK)      sIdx[tid] = IDX[tid];
  if (tid < BB * TOPK) sW[tid]   = W[tid];
  __syncthreads();

  size_t gid = (size_t)blockIdx.x * 256 + tid;
  int    c4  = gid & 127;
  size_t bl  = gid >> 7;
  int    l   = (int)(bl & LMASK);
  int    b   = (int)(bl >> 12);

  const float4* vb = (const float4*)V + (size_t)b * LL * 128;
  float4 o = make_float4(0.f, 0.f, 0.f, 0.f);
#pragma unroll
  for (int j = 0; j < TOPK; ++j) {
    int ls = (l + sIdx[j]) & LMASK;
    float wj = sW[b * TOPK + j];
    float4 p = vb[(size_t)ls * 128 + c4];
    o.x += wj * p.x; o.y += wj * p.y; o.z += wj * p.z; o.w += wj * p.w;
  }
  ((float4*)out)[gid] = o;
}

extern "C" void kernel_launch(void* const* d_in, const int* in_sizes, int n_in,
                              void* d_out, int out_size, void* d_ws, size_t ws_size,
                              hipStream_t stream) {
  const float* q = (const float*)d_in[0];
  const float* k = (const float*)d_in[1];
  const float* v = (const float*)d_in[2];
  // attn_mask (d_in[3]) is a scalar 0 in the reference -> unused

  float* C   = (float*)d_ws;
  int*   idx = (int*)((char*)d_ws + (size_t)BB * LL * sizeof(float));
  float* w   = (float*)((char*)d_ws + (size_t)BB * LL * sizeof(float) + 64);

  zero_ws<<<(BB * LL) / 256, 256, 0, stream>>>(C);

  const size_t NEEDWS = (size_t)(1u << 20) + 2ull * BB * LL * HE * sizeof(_Float16);
  if (ws_size >= NEEDWS) {
    _Float16* Qh = (_Float16*)((char*)d_ws + (1u << 20));
    _Float16* Kh = Qh + (size_t)BB * LL * HE;
    const int n4 = BB * LL * HE / 4;                         // 4,194,304 float4s
    cvt_to_f16<<<n4 / 256, 256, 0, stream>>>(q, Qh, n4);
    cvt_to_f16<<<n4 / 256, 256, 0, stream>>>(k, Kh, n4);
    corr_wmma_async<<<BB * (LL / QROWS), 256, 0, stream>>>(Qh, Kh, C);
  } else {
    corr_wmma_f32<<<BB * (LL / QROWS), 256, 0, stream>>>(q, k, C);
  }

  topk_softmax<<<1, 256, 0, stream>>>(C, idx, w);
  delay_agg<<<(BB * LL * HE / 4) / 256, 256, 0, stream>>>(v, w, idx, (float*)d_out);
}
